// NonLocalBlock_31250182046253
// MI455X (gfx1250) — matrix operand order repaired and checked by reading, not compiled, not run on hardware
//
#include <hip/hip_runtime.h>
#include <hip/hip_bf16.h>

// ---------------------------------------------------------------------------
// Non-local block (self-attention) for MI455X / gfx1250, wave32 + WMMA.
// B=4, C=256, H=W=64 -> HW=4096, INTER=128.
// Flash-attention streaming softmax: never materializes the 4096x4096 matrix.
// Matmuls: v_wmma_f32_16x16x32_f16 (f16 inputs, f32 accumulation).
// v2: 64-key slabs (half the shuffle cost/key), 8 waves/workgroup sharing
//     phi/g slabs through the WGP cache (8x less L2 traffic), prefetch.
// ---------------------------------------------------------------------------

#define NB    4
#define NC    256
#define NHW   4096
#define NI    128
#define BN_EPS 1e-5f

#define QTW   8     // query tiles (waves) per workgroup
#define KT    64    // key slab per iteration

typedef __attribute__((ext_vector_type(16))) _Float16 v16h;
typedef __attribute__((ext_vector_type(8)))  _Float16 v8h;
typedef __attribute__((ext_vector_type(8)))  float    v8f;
typedef __attribute__((ext_vector_type(4)))  float    v4f;

static __device__ inline v16h join8(v8h lo, v8h hi) {
  return __builtin_shufflevector(lo, hi, 0,1,2,3,4,5,6,7,8,9,10,11,12,13,14,15);
}

static __device__ inline float redmax16(float v) {
  // reduce across 16-lane halves (xor masks < 16 keep halves separate)
  v = fmaxf(v, __shfl_xor(v, 1, 32));
  v = fmaxf(v, __shfl_xor(v, 2, 32));
  v = fmaxf(v, __shfl_xor(v, 4, 32));
  v = fmaxf(v, __shfl_xor(v, 8, 32));
  return v;
}
static __device__ inline float redsum16(float v) {
  v += __shfl_xor(v, 1, 32);
  v += __shfl_xor(v, 2, 32);
  v += __shfl_xor(v, 4, 32);
  v += __shfl_xor(v, 8, 32);
  return v;
}

// ---------------------------------------------------------------------------
// Kernel 1: fused theta/phi/g 1x1 convolutions. fp32 math, f16 outputs.
// theta stored pixel-major  [b][pix][ic]   (WMMA A-fragment: K contiguous)
// phi   stored pixel-major  [b][pix][ic]   (WMMA B-fragment: K contiguous)
// g     stored channel-major[b][ic][pix]   (WMMA A-fragment: K contiguous)
// ---------------------------------------------------------------------------
__global__ __launch_bounds__(128)
void proj_kernel(const float* __restrict__ x,
                 const float* __restrict__ tw, const float* __restrict__ tb,
                 const float* __restrict__ pw, const float* __restrict__ pb,
                 const float* __restrict__ gw, const float* __restrict__ gbias,
                 _Float16* __restrict__ theta_pm,
                 _Float16* __restrict__ phi_cm,
                 _Float16* __restrict__ g_cm) {
  __shared__ __align__(16) float xv[NC];
  const int bp  = blockIdx.x;          // b*HW + pix
  const int b   = bp >> 12;
  const int pix = bp & (NHW - 1);
  const int t   = threadIdx.x;         // 0..127  (output channel)

  const float* xb = x + ((size_t)b * NC) * NHW + pix;
  xv[t]       = xb[(size_t)t * NHW];
  xv[t + 128] = xb[(size_t)(t + 128) * NHW];
  __syncthreads();

  float a0 = tb[t], a1 = pb[t], a2 = gbias[t];
  const float* w0 = tw + (size_t)t * NC;
  const float* w1 = pw + (size_t)t * NC;
  const float* w2 = gw + (size_t)t * NC;
#pragma unroll 4
  for (int c = 0; c < NC; c += 4) {
    const v4f xc = *(const v4f*)&xv[c];
    const v4f u0 = *(const v4f*)&w0[c];
    const v4f u1 = *(const v4f*)&w1[c];
    const v4f u2 = *(const v4f*)&w2[c];
#pragma unroll
    for (int k = 0; k < 4; ++k) {
      a0 += u0[k] * xc[k];
      a1 += u1[k] * xc[k];
      a2 += u2[k] * xc[k];
    }
  }
  theta_pm[(size_t)bp * NI + t]            = (_Float16)a0;
  phi_cm[(size_t)bp * NI + t]              = (_Float16)a1;
  g_cm  [((size_t)b * NI + t) * NHW + pix] = (_Float16)a2;
}

// ---------------------------------------------------------------------------
// Kernel 2: streaming-softmax attention.
// Workgroup = 8 waves (one WGP); wave w owns 16 queries; all 8 waves walk the
// same 64-key slab in lockstep (__syncthreads) so phi/g fetches are served
// out of the WGP cache after the first wave touches them.
// Per slab & wave:  16 S-WMMAs (K=128 in 4 chunks x 4 key sub-tiles),
//                   online softmax (shfl_xor row reductions),
//                   P^T staged to LDS f16 (C-layout -> B-layout),
//                   16 Y-WMMAs (8 channel tiles x 2 key chunks of 32).
// ---------------------------------------------------------------------------
__global__ __launch_bounds__(QTW * 32)
void attn_kernel(const _Float16* __restrict__ theta_pm,
                 const _Float16* __restrict__ phi_cm,
                 const _Float16* __restrict__ g_cm,
                 float* __restrict__ y_pm) {
  __shared__ __align__(16) _Float16 Pt[QTW][KT * 16];  // per-wave P^T [key][q]
  __shared__ float sc[QTW][16];                        // per-wave q broadcast

  const int lane = threadIdx.x & 31;
  const int wid  = threadIdx.x >> 5;            // wave id 0..7
  const int b    = blockIdx.x >> 5;             // 32 blocks per batch
  const int q0   = (((blockIdx.x & 31) * QTW) + wid) << 4;
  const int lm   = lane & 15;                   // = N index of C fragments
  const int lh   = lane >> 4;                   // lane half: K sub-block select
  const int koff = lh * 8;

  // --- preload theta A-fragments (16 queries x 128 channels), resident ----
  const _Float16* trow = theta_pm + ((size_t)(b * NHW) + q0 + lm) * NI;
  v16h thA[4];
#pragma unroll
  for (int kc = 0; kc < 4; ++kc) {
    v8h lo = *(const v8h*)(trow + kc * 32 + koff);       // K = lh*8 + 0..7
    v8h hi = *(const v8h*)(trow + kc * 32 + 16 + koff);  // K = 16 + lh*8 + 0..7
    thA[kc] = join8(lo, hi);
  }

  float mrow[8], lrow[8];
  v8f acc[8];
#pragma unroll
  for (int r = 0; r < 8; ++r) { mrow[r] = -1e30f; lrow[r] = 0.f; }
#pragma unroll
  for (int ct = 0; ct < 8; ++ct)
#pragma unroll
    for (int r = 0; r < 8; ++r) acc[ct][r] = 0.f;

  const _Float16* phib = phi_cm + (size_t)b * NI * NHW;
  const _Float16* gbp  = g_cm   + (size_t)b * NI * NHW;
  _Float16* ptw = &Pt[wid][0];
  float*    scw = &sc[wid][0];

  for (int kt = 0; kt < NHW; kt += KT) {
    __syncthreads();  // keep the 8 waves on the same slab for WGP$ reuse

    // ---------------- logits S (16 queries x 64 keys), f32 accum ----------
    v8f s[4];
#pragma unroll
    for (int j = 0; j < 4; ++j)
#pragma unroll
      for (int r = 0; r < 8; ++r) s[j][r] = 0.f;

#pragma unroll
    for (int kc = 0; kc < 4; ++kc) {
      // B fragment: lane = key pixel (N), 16 channels-in-chunk (K)
      const _Float16* prow = phib + (size_t)kt * NI + kc * 32;
      if (kt + KT < NHW) __builtin_prefetch(prow + KT * NI, 0, 0);
#pragma unroll
      for (int j = 0; j < 4; ++j) {
        v16h bp = join8(*(const v8h*)(prow + (j * 16 + lm) * NI + koff),
                        *(const v8h*)(prow + (j * 16 + lm) * NI + 16 + koff));
        s[j] = __builtin_amdgcn_wmma_f32_16x16x32_f16(false, thA[kc], false, bp,
                                                      (short)0, s[j],
                                                      false, false);
      }
    }

    // ---------------- online softmax over this 64-key slab ----------------
    float alpha[8];
#pragma unroll
    for (int r = 0; r < 8; ++r) {
      float tm = fmaxf(fmaxf(s[0][r], s[1][r]), fmaxf(s[2][r], s[3][r]));
      tm = redmax16(tm);
      const float mn = fmaxf(mrow[r], tm);
      alpha[r] = __expf(mrow[r] - mn);
      float rs = 0.f;
#pragma unroll
      for (int j = 0; j < 4; ++j) {
        s[j][r] = __expf(s[j][r] - mn);   // exp in place (becomes P)
        rs += s[j][r];
      }
      rs = redsum16(rs);
      lrow[r] = lrow[r] * alpha[r] + rs;
      mrow[r] = mn;
    }
    // broadcast per-query rescale alpha: row (r + lh*8) stats are replicated
    // in one 16-lane half; lane lm==r of each half publishes to LDS.
#pragma unroll
    for (int r = 0; r < 8; ++r)
      if (lm == r) scw[lh * 8 + r] = alpha[r];

    // store P^T (64 keys x 16 queries) f16: lane owns (key=j*16+lm, q=lh*8..+7)
#pragma unroll
    for (int j = 0; j < 4; ++j) {
      v8h p;
#pragma unroll
      for (int r = 0; r < 8; ++r) p[r] = (_Float16)s[j][r];
      *(v8h*)&ptw[(j * 16 + lm) * 16 + lh * 8] = p;
    }

    const float al = scw[lm];  // rescale factor for this lane's query column
#pragma unroll
    for (int ct = 0; ct < 8; ++ct)
#pragma unroll
      for (int r = 0; r < 8; ++r) acc[ct][r] *= al;

    // ---------------- Y^T update: 2 key chunks x 8 channel tiles ----------
#pragma unroll
    for (int cb = 0; cb < 2; ++cb) {
      // B fragment of P^T: lane = query (N), 16 keys-in-chunk (K)
      v16h pB;
#pragma unroll
      for (int s = 0; s < 16; ++s)
        pB[s] = ptw[(cb * 32 + (s < 8 ? koff + s : 8 + koff + s)) * 16 + lm];
#pragma unroll
      for (int ct = 0; ct < 8; ++ct) {
        const _Float16* grow = gbp + (size_t)(ct * 16 + lm) * NHW + kt + cb * 32;
        v16h gA = join8(*(const v8h*)(grow + koff),
                        *(const v8h*)(grow + 16 + koff));
        acc[ct] = __builtin_amdgcn_wmma_f32_16x16x32_f16(false, gA, false, pB,
                                                         (short)0, acc[ct],
                                                         false, false);
      }
    }
  }

  // ---------------- final 1/l normalization + store ------------------------
#pragma unroll
  for (int r = 0; r < 8; ++r)
    if (lm == r) scw[lh * 8 + r] = 1.f / lrow[r];
  const float il = scw[lm];

  float* yrow = y_pm + ((size_t)(b * NHW) + q0 + lm) * NI;
#pragma unroll
  for (int ct = 0; ct < 8; ++ct) {
    v4f o0, o1;
#pragma unroll
    for (int r = 0; r < 4; ++r) { o0[r] = acc[ct][r] * il; o1[r] = acc[ct][4 + r] * il; }
    *(v4f*)(yrow + ct * 16 + lh * 8)     = o0;   // channels ct*16 + lh*8 + 0..3
    *(v4f*)(yrow + ct * 16 + lh * 8 + 4) = o1;   // channels ct*16 + lh*8 + 4..7
  }
}

// ---------------------------------------------------------------------------
// Kernel 3: W 1x1 conv (128 -> 256), fp32.  Block = one pixel, thread = out ch.
// wy stored channel-major [b][c][pix] (BN reduction friendly).
// ---------------------------------------------------------------------------
__global__ __launch_bounds__(256)
void wproj_kernel(const float* __restrict__ y_pm,
                  const float* __restrict__ Ww, const float* __restrict__ Wb,
                  float* __restrict__ wy) {
  __shared__ __align__(16) float yv[NI];
  const int bp  = blockIdx.x;
  const int b   = bp >> 12;
  const int pix = bp & (NHW - 1);
  const int t   = threadIdx.x;  // 0..255
  if (t < NI) yv[t] = y_pm[(size_t)bp * NI + t];
  __syncthreads();

  float a = Wb[t];
  const float* w = Ww + (size_t)t * NI;
#pragma unroll 4
  for (int i = 0; i < NI; i += 4) {
    const v4f yc = *(const v4f*)&yv[i];
    const v4f wc = *(const v4f*)&w[i];
#pragma unroll
    for (int k = 0; k < 4; ++k) a += wc[k] * yc[k];
  }
  wy[((size_t)b * NC + t) * NHW + pix] = a;
}

// ---------------------------------------------------------------------------
// Kernel 4: per-channel batch statistics (training-mode BN).
// ---------------------------------------------------------------------------
__global__ __launch_bounds__(256)
void bnstat_kernel(const float* __restrict__ wy,
                   float* __restrict__ meanv, float* __restrict__ invv) {
  __shared__ float r1[256], r2[256];
  const int c = blockIdx.x;
  const int t = threadIdx.x;
  float s1 = 0.f, s2 = 0.f;
  for (int i = t; i < NB * NHW; i += 256) {
    const int bb = i >> 12, p = i & (NHW - 1);
    const float v = wy[((size_t)bb * NC + c) * NHW + p];
    s1 += v; s2 += v * v;
  }
  r1[t] = s1; r2[t] = s2;
  __syncthreads();
  for (int s = 128; s > 0; s >>= 1) {
    if (t < s) { r1[t] += r1[t + s]; r2[t] += r2[t + s]; }
    __syncthreads();
  }
  if (t == 0) {
    const float n = (float)(NB * NHW);
    const float m = r1[0] / n;
    const float var = r2[0] / n - m * m;
    meanv[c] = m;
    invv[c]  = rsqrtf(var + BN_EPS);
  }
}

// ---------------------------------------------------------------------------
// Kernel 5: BN apply + residual add.
// ---------------------------------------------------------------------------
__global__ __launch_bounds__(256)
void bnout_kernel(const float* __restrict__ wy, const float* __restrict__ x,
                  const float* __restrict__ gamma, const float* __restrict__ beta,
                  const float* __restrict__ meanv, const float* __restrict__ invv,
                  float* __restrict__ out, int n) {
  const int idx = blockIdx.x * blockDim.x + threadIdx.x;
  if (idx >= n) return;
  const int c = (idx >> 12) & (NC - 1);
  out[idx] = gamma[c] * (wy[idx] - meanv[c]) * invv[c] + beta[c] + x[idx];
}

// ---------------------------------------------------------------------------
// Launch.  Workspace (d_ws) layout, ~36 MB total:
//   theta_pm f16 4MB | phi_cm f16 4MB | g_cm f16 4MB | y_pm f32 8MB |
//   wy f32 16MB | mean 1KB | inv 1KB
// ---------------------------------------------------------------------------
extern "C" void kernel_launch(void* const* d_in, const int* in_sizes, int n_in,
                              void* d_out, int out_size, void* d_ws, size_t ws_size,
                              hipStream_t stream) {
  const float* x     = (const float*)d_in[0];
  const float* tw    = (const float*)d_in[1];
  const float* tb    = (const float*)d_in[2];
  const float* pw    = (const float*)d_in[3];
  const float* pb    = (const float*)d_in[4];
  const float* gw    = (const float*)d_in[5];
  const float* gbias = (const float*)d_in[6];
  const float* Ww    = (const float*)d_in[7];
  const float* Wb    = (const float*)d_in[8];
  const float* gamma = (const float*)d_in[9];
  const float* beta  = (const float*)d_in[10];
  float* out = (float*)d_out;

  const size_t nProj = (size_t)NB * NHW * NI;   // 2M elements
  _Float16* theta_pm = (_Float16*)d_ws;
  _Float16* phi_cm   = theta_pm + nProj;
  _Float16* g_cm     = phi_cm + nProj;
  float*    y_pm     = (float*)(g_cm + nProj);  // 12MB offset, 16B aligned
  float*    wy       = y_pm + nProj;
  float*    meanv    = wy + (size_t)NB * NC * NHW;
  float*    invv     = meanv + NC;

  proj_kernel<<<NB * NHW, 128, 0, stream>>>(x, tw, tb, pw, pb, gw, gbias,
                                            theta_pm, phi_cm, g_cm);
  attn_kernel<<<NB * (NHW / 16) / QTW, QTW * 32, 0, stream>>>(theta_pm, phi_cm,
                                                              g_cm, y_pm);
  wproj_kernel<<<NB * NHW, 256, 0, stream>>>(y_pm, Ww, Wb, wy);
  bnstat_kernel<<<NC, 256, 0, stream>>>(wy, meanv, invv);
  const int n = NB * NC * NHW;
  bnout_kernel<<<(n + 255) / 256, 256, 0, stream>>>(wy, x, gamma, beta, meanv,
                                                    invv, out, n);
}
